// MultiheadLinearSelfAttentionKernalization_27865747817033
// MI455X (gfx1250) — matrix operand order repaired and checked by reading, not compiled, not run on hardware
//
#include <hip/hip_runtime.h>

typedef unsigned short u16;
typedef __attribute__((ext_vector_type(16))) __bf16        v16bf;
typedef __attribute__((ext_vector_type(8)))  float         v8f;
typedef __attribute__((ext_vector_type(4)))  unsigned int  v4u;
typedef __attribute__((ext_vector_type(4)))  float         v4f;
typedef __attribute__((ext_vector_type(4)))  unsigned int  u32x4;
typedef __attribute__((ext_vector_type(8)))  int           i32x8;
typedef __attribute__((ext_vector_type(4)))  int           i32x4;

#define B_     2
#define S_     2048
#define D_     1024
#define H_     16
#define DEPTH_ 64
#define ROWS   (B_ * S_)   // 4096
#define EPS_   1e-6f

#if __has_builtin(__builtin_amdgcn_tensor_load_to_lds)
#define HAVE_TDM 1
#else
#define HAVE_TDM 0
#endif

__device__ __forceinline__ u16 f2bf(float f) {
  unsigned int u = __float_as_uint(f);
  u += 0x7FFFu + ((u >> 16) & 1u);     // round-to-nearest-even
  return (u16)(u >> 16);
}
__device__ __forceinline__ float bf2f(u16 h) {
  return __uint_as_float(((unsigned int)h) << 16);
}
__device__ __forceinline__ v8f v8f_zero() {
  v8f z = {0.f, 0.f, 0.f, 0.f, 0.f, 0.f, 0.f, 0.f};
  return z;
}

union FragBF { v16bf v; v4u q[2]; };
union Pack8  { u16 o[8]; v4u q; };

#if HAVE_TDM
// ---------------------------------------------------------------------------
// Tensor Data Mover helpers (CDNA5 ISA ch.8).
// Group1 (data size, tensor/tile dims, row stride) is loop-invariant and
// built once; per-issue work is just the group0 address words.
// ---------------------------------------------------------------------------
__device__ __forceinline__ i32x8 tdm_make_g1(unsigned tensor_w, unsigned tensor_h,
                                             unsigned tile_w, unsigned tile_h) {
  i32x8 g1;
  g1[0] = (int)(1u << 16);                                    // data_size=1 (2B)
  g1[1] = (int)((tensor_w & 0xFFFFu) << 16);                  // tensor_dim0 lo
  g1[2] = (int)(((tensor_w >> 16) & 0xFFFFu) | ((tensor_h & 0xFFFFu) << 16));
  g1[3] = (int)(((tensor_h >> 16) & 0xFFFFu) | ((tile_w & 0xFFFFu) << 16));
  g1[4] = (int)(tile_h & 0xFFFFu);                            // tile_dim1, dim2=0
  g1[5] = (int)tensor_w;                                      // dim0_stride lo
  g1[6] = 0;
  g1[7] = 0;
  return g1;
}

__device__ __forceinline__ void tdm_issue(const u16* gbase, unsigned lds_off, i32x8 g1) {
  unsigned long long ga = (unsigned long long)(uintptr_t)gbase;
  u32x4 g0;
  g0.x = 1u;                                                  // count=1 valid
  g0.y = lds_off;                                             // lds_addr
  g0.z = (unsigned)(ga & 0xFFFFFFFFu);                        // gaddr[31:0]
  g0.w = (unsigned)((ga >> 32) & 0x01FFFFFFu) | (2u << 30);   // gaddr[56:32]|type=2
  i32x4 z4 = {0, 0, 0, 0};
#if defined(__clang_major__) && (__clang_major__ >= 23)
  i32x8 z8 = {0, 0, 0, 0, 0, 0, 0, 0};
  __builtin_amdgcn_tensor_load_to_lds(g0, g1, z4, z4, z8, 0);
#else
  __builtin_amdgcn_tensor_load_to_lds(g0, g1, z4, z4, 0);
#endif
}

__device__ __forceinline__ void tdm_wait() {
#if __has_builtin(__builtin_amdgcn_s_wait_tensorcnt)
  __builtin_amdgcn_s_wait_tensorcnt(0);
#else
  asm volatile("s_wait_tensorcnt 0x0" ::: "memory");
#endif
}
#endif  // HAVE_TDM

// ---------------------------------------------------------------------------
// fp32 -> bf16 conversion, 4 elements/thread (row-major copy, used for x)
// ---------------------------------------------------------------------------
__global__ void cvt_bf16_kernel(const float* __restrict__ src, u16* __restrict__ dst, int n4) {
  int i = blockIdx.x * blockDim.x + threadIdx.x;
  if (i >= n4) return;
  v4f f = *(const v4f*)(src + (size_t)i * 4);
  unsigned long long pk =  (unsigned long long)f2bf(f.x)
                        | ((unsigned long long)f2bf(f.y) << 16)
                        | ((unsigned long long)f2bf(f.z) << 32)
                        | ((unsigned long long)f2bf(f.w) << 48);
  *(unsigned long long*)(dst + (size_t)i * 4) = pk;
}

// ---------------------------------------------------------------------------
// fp32 [K][N] -> bf16 transposed [N][K] (one-time weight prep, tiled in LDS)
// ---------------------------------------------------------------------------
__global__ void cvt_transpose_kernel(const float* __restrict__ W, u16* __restrict__ Wt) {
  __shared__ float tile[32][33];
  const int k0 = blockIdx.x * 32;
  const int n0 = blockIdx.y * 32;
  const int tx = threadIdx.x & 31, ty = threadIdx.x >> 5;   // 32 x 8
  #pragma unroll
  for (int p = 0; p < 4; ++p) {
    int r = ty + p * 8;
    tile[r][tx] = W[(size_t)(k0 + r) * D_ + n0 + tx];
  }
  __syncthreads();
  #pragma unroll
  for (int p = 0; p < 4; ++p) {
    int r = ty + p * 8;
    Wt[(size_t)(n0 + r) * D_ + k0 + tx] = f2bf(tile[tx][r]);
  }
}

// ---------------------------------------------------------------------------
// 64x64 block-tile bf16 GEMM core: C = A[m0:,k] * BT[n0:,k]^T
// TDM path: double-buffered ping-pong.  Wave 0 waits TENSORcnt for buffer
// `cur`, the barrier publishes it, then the NEXT tile pair is DMA'd into
// buffer `cur^1` while all 4 waves run the WMMAs on `cur` -> DMA/compute
// overlap with one pair in flight.  Fallback: manual b128 staging.
// ---------------------------------------------------------------------------
__device__ __forceinline__ void gemm_block_64x64(
    const u16* __restrict__ A, const u16* __restrict__ BT,
    int m0, int n0, int K,
    u16 (*As)[64][32], u16 (*Bs)[64][32],
    v8f acc[2][2])
{
  const int t    = threadIdx.x;                  // 0..127
  const int lane = t & 31;
  const int wv   = t >> 5;                       // wave 0..3
  const int wm   = wv >> 1, wn = wv & 1;
  const int lr   = lane & 15, hi = lane >> 4;

#if HAVE_TDM
  // LDS is written only by the TDM; keep a (never-executed) VALU store so the
  // compiler cannot undef-fold the ds_loads of DMA-written tiles.
  if (blockIdx.x == 0x7FFFFFFFu) { As[0][0][0] = (u16)K; Bs[0][0][0] = (u16)K; }
  const unsigned lofA[2] = { (unsigned)(uintptr_t)&As[0][0][0],
                             (unsigned)(uintptr_t)&As[1][0][0] };
  const unsigned lofB[2] = { (unsigned)(uintptr_t)&Bs[0][0][0],
                             (unsigned)(uintptr_t)&Bs[1][0][0] };
  const i32x8 g1 = tdm_make_g1(D_, 65536u, 32u, 64u);
  const u16* Abase = A  + (size_t)m0 * D_;
  const u16* Bbase = BT + (size_t)n0 * D_;
  if (wv == 0) {                                 // prologue: fill buffer 0
    tdm_issue(Abase, lofA[0], g1);
    tdm_issue(Bbase, lofB[0], g1);
  }
  for (int k0 = 0; k0 < K; k0 += 32) {
    const int cur = (k0 >> 5) & 1;
    if (wv == 0) tdm_wait();                     // buffer `cur` complete
    __syncthreads();                             // publish cur; cur^1 drained
    if (wv == 0 && k0 + 32 < K) {                // async fill of cur^1
      tdm_issue(Abase + k0 + 32, lofA[cur ^ 1], g1);
      tdm_issue(Bbase + k0 + 32, lofB[cur ^ 1], g1);
    }
    #pragma unroll
    for (int i = 0; i < 2; ++i) {
      FragBF a;
      int row = (wm * 2 + i) * 16 + lr;
      a.q[0] = *(const v4u*)&As[cur][row][hi * 8];
      a.q[1] = *(const v4u*)&As[cur][row][hi * 8 + 16];
      #pragma unroll
      for (int j = 0; j < 2; ++j) {
        FragBF b;
        int nrow = (wn * 2 + j) * 16 + lr;
        b.q[0] = *(const v4u*)&Bs[cur][nrow][hi * 8];
        b.q[1] = *(const v4u*)&Bs[cur][nrow][hi * 8 + 16];
        acc[i][j] = __builtin_amdgcn_wmma_f32_16x16x32_bf16(
            false, a.v, false, b.v, (short)0, acc[i][j], false, false);
      }
    }
  }
#else
  const int r = t >> 2, c = (t & 3) * 8;
  for (int k0 = 0; k0 < K; k0 += 32) {
    __syncthreads();
    *(v4u*)&As[0][r][c]      = *(const v4u*)&A [(size_t)(m0 + r)      * D_ + k0 + c];
    *(v4u*)&As[0][r + 32][c] = *(const v4u*)&A [(size_t)(m0 + r + 32) * D_ + k0 + c];
    *(v4u*)&Bs[0][r][c]      = *(const v4u*)&BT[(size_t)(n0 + r)      * D_ + k0 + c];
    *(v4u*)&Bs[0][r + 32][c] = *(const v4u*)&BT[(size_t)(n0 + r + 32) * D_ + k0 + c];
    if (k0 + 32 < K) {
      __builtin_prefetch(&A [(size_t)(m0 + r) * D_ + k0 + 32 + c], 0, 1);
      __builtin_prefetch(&BT[(size_t)(n0 + r) * D_ + k0 + 32 + c], 0, 1);
    }
    __syncthreads();
    #pragma unroll
    for (int i = 0; i < 2; ++i) {
      FragBF a;
      int row = (wm * 2 + i) * 16 + lr;
      a.q[0] = *(const v4u*)&As[0][row][hi * 8];
      a.q[1] = *(const v4u*)&As[0][row][hi * 8 + 16];
      #pragma unroll
      for (int j = 0; j < 2; ++j) {
        FragBF b;
        int nrow = (wn * 2 + j) * 16 + lr;
        b.q[0] = *(const v4u*)&Bs[0][nrow][hi * 8];
        b.q[1] = *(const v4u*)&Bs[0][nrow][hi * 8 + 16];
        acc[i][j] = __builtin_amdgcn_wmma_f32_16x16x32_bf16(
            false, a.v, false, b.v, (short)0, acc[i][j], false, false);
      }
    }
  }
#endif
}

// ---------------------------------------------------------------------------
// QKV projection: P = x @ W{q,k,v} + b.
//   z=0 (Q): elu+1, row-major bf16   (A-side of the attn GEMM)
//   z=1 (K): elu+1, row-major bf16   (input to knorm)
//   z=2 (V): transposed bf16 Vt[bh*64+d][s] via packed b128 stores
// ---------------------------------------------------------------------------
__global__ void qkv_gemm_kernel(
    const u16* __restrict__ xb,
    const u16* __restrict__ Wqt, const u16* __restrict__ Wkt, const u16* __restrict__ Wvt,
    const float* __restrict__ bq, const float* __restrict__ bk, const float* __restrict__ bv,
    u16* __restrict__ Qb, u16* __restrict__ Kb, u16* __restrict__ Vt)
{
  __shared__ u16 As[2][64][32];
  __shared__ u16 Bs[2][64][32];
  const int z = blockIdx.z;
  const u16*   W   = (z == 0) ? Wqt : (z == 1) ? Wkt : Wvt;
  const float* bia = (z == 0) ? bq  : (z == 1) ? bk  : bv;
  const int m0 = blockIdx.y * 64, n0 = blockIdx.x * 64;

  v8f acc[2][2];
  acc[0][0] = v8f_zero(); acc[0][1] = v8f_zero();
  acc[1][0] = v8f_zero(); acc[1][1] = v8f_zero();
  gemm_block_64x64(xb, W, m0, n0, D_, As, Bs, acc);

  const int t = threadIdx.x, lane = t & 31, wv = t >> 5;
  const int wm = wv >> 1, wn = wv & 1, lr = lane & 15, hi = lane >> 4;
  const int b = m0 >> 11;                 // batch index (64 | 2048)
  #pragma unroll
  for (int i = 0; i < 2; ++i)
    #pragma unroll
    for (int j = 0; j < 2; ++j) {
      int col = n0 + (wn * 2 + j) * 16 + lr;
      float bcol = bia[col];
      if (z == 2) {                       // V -> transposed, packed store
        Pack8 pk;
        #pragma unroll
        for (int r = 0; r < 8; ++r) pk.o[r] = f2bf(acc[i][j][r] + bcol);
        int h = col >> 6, d = col & 63;
        int sb = (m0 & (S_ - 1)) + (wm * 2 + i) * 16 + hi * 8;
        *(v4u*)&Vt[((size_t)(b * H_ + h) * 64 + d) * S_ + sb] = pk.q;
      } else {                            // Q / K: elu+1, row-major
        u16* dst = (z == 0) ? Qb : Kb;
        #pragma unroll
        for (int r = 0; r < 8; ++r) {
          int row = m0 + (wm * 2 + i) * 16 + hi * 8 + r;
          float v = acc[i][j][r] + bcol;
          v = (v > 0.f) ? (v + 1.f) : __expf(v);   // elu(v)+1
          dst[(size_t)row * D_ + col] = f2bf(v);
        }
      }
    }
}

// ---------------------------------------------------------------------------
// K per-(token,head) L2 normalization (eps inside sqrt), row-major Kb in,
// transposed Kt[bh*64+d][s] out (lane-consecutive tokens -> coalesced lines).
// ---------------------------------------------------------------------------
__global__ void knorm_kernel(const u16* __restrict__ Kb, u16* __restrict__ Kt) {
  int tid = blockIdx.x * blockDim.x + threadIdx.x;   // ROWS*H_ = 65536
  int sg = tid & (ROWS - 1);                         // global token row
  int h  = tid >> 12;                                // head
  int b  = sg >> 11, s = sg & (S_ - 1);
  const u16* p = Kb + (size_t)sg * D_ + h * DEPTH_;
  v4u buf[8];
  float ss = 0.f;
  #pragma unroll
  for (int c = 0; c < 8; ++c) {
    buf[c] = *(const v4u*)&p[c * 8];
    const u16* w = (const u16*)&buf[c];
    #pragma unroll
    for (int j = 0; j < 8; ++j) { float v = bf2f(w[j]); ss += v * v; }
  }
  float inv = rsqrtf(ss + EPS_);
  size_t base = (size_t)(b * H_ + h) * 64 * S_ + s;
  #pragma unroll
  for (int c = 0; c < 8; ++c) {
    const u16* w = (const u16*)&buf[c];
    #pragma unroll
    for (int j = 0; j < 8; ++j)
      Kt[base + (size_t)(c * 8 + j) * S_] = f2bf(bf2f(w[j]) * inv);
  }
}

// ---------------------------------------------------------------------------
// KV = K^T V per (b,h): 64x64 f32 over S via WMMA, zero LDS.
// Kt/Vt are [d][s] = fragment-major; every fragment is two aligned b128 loads.
// ---------------------------------------------------------------------------
__global__ void kv_kernel(const u16* __restrict__ Kt, const u16* __restrict__ Vt,
                          u16* __restrict__ KVt) {
  const int bh = blockIdx.x;            // B_*H_
  const int t = threadIdx.x;            // 0..255 = 8 waves
  const int lane = t & 31, wv = t >> 5;
  const int mt = wv >> 1, ntb = (wv & 1) * 2;
  const int lr = lane & 15, hi = lane >> 4;
  v8f acc[2];
  acc[0] = v8f_zero(); acc[1] = v8f_zero();

  const u16* Arow  = Kt + ((size_t)bh * 64 + mt * 16 + lr) * S_;
  const u16* Brow0 = Vt + ((size_t)bh * 64 + (ntb + 0) * 16 + lr) * S_;
  const u16* Brow1 = Vt + ((size_t)bh * 64 + (ntb + 1) * 16 + lr) * S_;

  for (int s0 = 0; s0 < S_; s0 += 32) {
    FragBF a, b0, b1;
    a.q[0]  = *(const v4u*)&Arow [s0 + hi * 8];
    a.q[1]  = *(const v4u*)&Arow [s0 + hi * 8 + 16];
    b0.q[0] = *(const v4u*)&Brow0[s0 + hi * 8];
    b0.q[1] = *(const v4u*)&Brow0[s0 + hi * 8 + 16];
    b1.q[0] = *(const v4u*)&Brow1[s0 + hi * 8];
    b1.q[1] = *(const v4u*)&Brow1[s0 + hi * 8 + 16];
    if (s0 + 128 < S_) {
      __builtin_prefetch(&Arow [s0 + 128], 0, 1);
      __builtin_prefetch(&Brow0[s0 + 128], 0, 1);
    }
    acc[0] = __builtin_amdgcn_wmma_f32_16x16x32_bf16(
        false, a.v, false, b0.v, (short)0, acc[0], false, false);
    acc[1] = __builtin_amdgcn_wmma_f32_16x16x32_bf16(
        false, a.v, false, b1.v, (short)0, acc[1], false, false);
  }
  #pragma unroll
  for (int j = 0; j < 2; ++j) {
    int n = (ntb + j) * 16 + lr;
    #pragma unroll
    for (int r = 0; r < 8; ++r) {
      int m = mt * 16 + hi * 8 + r;
      KVt[((size_t)bh * 64 + n) * 64 + m] = f2bf(acc[j][r]);
    }
  }
}

// ---------------------------------------------------------------------------
// attn = Q @ KV per head (M=4096 rows, K=N=64). Tiny K: straight from global.
// ---------------------------------------------------------------------------
__global__ void attn_kernel(const u16* __restrict__ Qb, const u16* __restrict__ KVt,
                            u16* __restrict__ attnb) {
  const int m0 = blockIdx.x * 64;
  const int h  = blockIdx.y;
  const int b  = m0 >> 11;
  const int t = threadIdx.x, lane = t & 31, wv = t >> 5;
  const int wm = wv >> 1, wn = wv & 1, lr = lane & 15, hi = lane >> 4;
  v8f acc[2][2];
  acc[0][0] = v8f_zero(); acc[0][1] = v8f_zero();
  acc[1][0] = v8f_zero(); acc[1][1] = v8f_zero();

  #pragma unroll
  for (int kt = 0; kt < 2; ++kt) {
    #pragma unroll
    for (int i = 0; i < 2; ++i) {
      FragBF a;
      const u16* ar = Qb + (size_t)(m0 + (wm * 2 + i) * 16 + lr) * D_ + h * DEPTH_ + kt * 32;
      a.q[0] = *(const v4u*)&ar[hi * 8];
      a.q[1] = *(const v4u*)&ar[hi * 8 + 16];
      #pragma unroll
      for (int j = 0; j < 2; ++j) {
        FragBF bb;
        const u16* br = KVt + ((size_t)(b * H_ + h) * 64 + (wn * 2 + j) * 16 + lr) * 64 + kt * 32;
        bb.q[0] = *(const v4u*)&br[hi * 8];
        bb.q[1] = *(const v4u*)&br[hi * 8 + 16];
        acc[i][j] = __builtin_amdgcn_wmma_f32_16x16x32_bf16(
            false, a.v, false, bb.v, (short)0, acc[i][j], false, false);
      }
    }
  }
  #pragma unroll
  for (int i = 0; i < 2; ++i)
    #pragma unroll
    for (int j = 0; j < 2; ++j) {
      int col = h * DEPTH_ + (wn * 2 + j) * 16 + lr;
      #pragma unroll
      for (int r = 0; r < 8; ++r) {
        int row = m0 + (wm * 2 + i) * 16 + hi * 8 + r;
        attnb[(size_t)row * D_ + col] = f2bf(acc[i][j][r]);
      }
    }
}

// ---------------------------------------------------------------------------
// Output projection: out = attn @ Wo + bo (f32 out for the LN pass)
// ---------------------------------------------------------------------------
__global__ void oproj_gemm_kernel(const u16* __restrict__ Ab, const u16* __restrict__ Wot,
                                  const float* __restrict__ bo, float* __restrict__ out) {
  __shared__ u16 As[2][64][32];
  __shared__ u16 Bs[2][64][32];
  const int m0 = blockIdx.y * 64, n0 = blockIdx.x * 64;
  v8f acc[2][2];
  acc[0][0] = v8f_zero(); acc[0][1] = v8f_zero();
  acc[1][0] = v8f_zero(); acc[1][1] = v8f_zero();
  gemm_block_64x64(Ab, Wot, m0, n0, D_, As, Bs, acc);

  const int t = threadIdx.x, lane = t & 31, wv = t >> 5;
  const int wm = wv >> 1, wn = wv & 1, lr = lane & 15, hi = lane >> 4;
  #pragma unroll
  for (int i = 0; i < 2; ++i)
    #pragma unroll
    for (int j = 0; j < 2; ++j) {
      int col = n0 + (wn * 2 + j) * 16 + lr;
      float bcol = bo[col];
      #pragma unroll
      for (int r = 0; r < 8; ++r) {
        int row = m0 + (wm * 2 + i) * 16 + hi * 8 + r;
        out[(size_t)row * D_ + col] = acc[i][j][r] + bcol;
      }
    }
}

// ---------------------------------------------------------------------------
// Residual + LayerNorm over last axis (var = E[y^2] - mu^2, eps in sqrt)
// ---------------------------------------------------------------------------
__global__ void ln_kernel(const float* __restrict__ x, const float* __restrict__ pout,
                          const float* __restrict__ gamma, const float* __restrict__ beta,
                          float* __restrict__ out) {
  __shared__ float s1[256];
  __shared__ float s2[256];
  const int row = blockIdx.x;
  const int t = threadIdx.x;
  float y[4];
  float sum = 0.f, sq = 0.f;
  #pragma unroll
  for (int c = 0; c < 4; ++c) {
    int idx = row * D_ + t + c * 256;
    y[c] = x[idx] + pout[idx];
    sum += y[c];
    sq += y[c] * y[c];
  }
  s1[t] = sum; s2[t] = sq;
  __syncthreads();
  for (int o = 128; o > 0; o >>= 1) {
    if (t < o) { s1[t] += s1[t + o]; s2[t] += s2[t + o]; }
    __syncthreads();
  }
  float mu  = s1[0] * (1.f / 1024.f);
  float var = s2[0] * (1.f / 1024.f) - mu * mu;
  float inv = rsqrtf(var + EPS_);
  #pragma unroll
  for (int c = 0; c < 4; ++c) {
    int col = t + c * 256;
    out[row * D_ + col] = (y[c] - mu) * inv * gamma[col] + beta[col];
  }
}

// ---------------------------------------------------------------------------
extern "C" void kernel_launch(void* const* d_in, const int* in_sizes, int n_in,
                              void* d_out, int out_size, void* d_ws, size_t ws_size,
                              hipStream_t stream) {
  (void)in_sizes; (void)n_in; (void)out_size; (void)ws_size;
  const float* x     = (const float*)d_in[0];
  const float* Wq    = (const float*)d_in[1];
  const float* bq    = (const float*)d_in[2];
  const float* Wk    = (const float*)d_in[3];
  const float* bk    = (const float*)d_in[4];
  const float* Wv    = (const float*)d_in[5];
  const float* bv    = (const float*)d_in[6];
  const float* Wo    = (const float*)d_in[7];
  const float* bo    = (const float*)d_in[8];
  const float* gamma = (const float*)d_in[9];
  const float* beta  = (const float*)d_in[10];
  float* out = (float*)d_out;

  char* ws = (char*)d_ws;
  size_t off = 0;
  auto alloc = [&](size_t bytes) -> void* {
    void* p = ws + off;
    off += (bytes + 255) & ~(size_t)255;
    return p;
  };
  u16* xb    = (u16*)alloc((size_t)ROWS * D_ * 2);
  u16* Wqt   = (u16*)alloc((size_t)D_ * D_ * 2);
  u16* Wkt   = (u16*)alloc((size_t)D_ * D_ * 2);
  u16* Wvt   = (u16*)alloc((size_t)D_ * D_ * 2);
  u16* Wot   = (u16*)alloc((size_t)D_ * D_ * 2);
  u16* Qb    = (u16*)alloc((size_t)ROWS * D_ * 2);
  u16* Kb    = (u16*)alloc((size_t)ROWS * D_ * 2);
  u16* Kt    = (u16*)alloc((size_t)ROWS * D_ * 2);
  u16* Vt    = (u16*)alloc((size_t)ROWS * D_ * 2);
  u16* KVt   = (u16*)alloc((size_t)B_ * H_ * 64 * 64 * 2);
  u16* attnb = (u16*)alloc((size_t)ROWS * D_ * 2);
  float* po  = (float*)alloc((size_t)ROWS * D_ * 4);

  // 1) x fp32->bf16 (row-major); weights fp32->bf16 transposed (one-time)
  cvt_bf16_kernel<<<(ROWS * D_ / 4) / 256, 256, 0, stream>>>(x, xb, ROWS * D_ / 4);
  dim3 tgrid(D_ / 32, D_ / 32);
  cvt_transpose_kernel<<<tgrid, 256, 0, stream>>>(Wq, Wqt);
  cvt_transpose_kernel<<<tgrid, 256, 0, stream>>>(Wk, Wkt);
  cvt_transpose_kernel<<<tgrid, 256, 0, stream>>>(Wv, Wvt);
  cvt_transpose_kernel<<<tgrid, 256, 0, stream>>>(Wo, Wot);

  // 2) QKV projections + bias + elu+1 (WMMA, double-buffered TDM tiles)
  qkv_gemm_kernel<<<dim3(D_ / 64, ROWS / 64, 3), 128, 0, stream>>>(
      xb, Wqt, Wkt, Wvt, bq, bk, bv, Qb, Kb, Vt);

  // 3) K per-head L2 normalization -> transposed Kt
  knorm_kernel<<<(ROWS * H_) / 256, 256, 0, stream>>>(Kb, Kt);

  // 4) KV = K^T V per (b,h)  (linear-attention associativity, LDS-free WMMA)
  kv_kernel<<<B_ * H_, 256, 0, stream>>>(Kt, Vt, KVt);

  // 5) attn = Q @ KV per head
  attn_kernel<<<dim3(ROWS / 64, H_), 128, 0, stream>>>(Qb, KVt, attnb);

  // 6) out-projection + bias (f32)
  oproj_gemm_kernel<<<dim3(D_ / 64, ROWS / 64), 128, 0, stream>>>(attnb, Wot, bo, po);

  // 7) residual + LayerNorm
  ln_kernel<<<ROWS, 256, 0, stream>>>(x, po, gamma, beta, out);
}
